// nondecayRNN_72464688218249
// MI455X (gfx1250) — compile-verified, matrix operand order checked by reading, not run on hardware
//
#include <hip/hip_runtime.h>

// ---------------------------------------------------------------------------
// nondecay RNN forward for MI455X (gfx1250, wave32, WMMA)
//   hiddens[b,t,:] = relu([x[b,t,:] | h_{t-1}] @ [W_in | Wrec]^T + b_in)
//   outs[b,t,:]    = softmax(hiddens[b,t,:] @ W_out^T + b_out)
//
// The input projection is folded into the recurrence as a single GEMM over
// concatenated K = I + H = 1152: one uniform fragment stream, no separate
// W_in loop for the compiler to hoist (fixes round-2 register spills).
// ---------------------------------------------------------------------------

typedef __attribute__((ext_vector_type(16))) __bf16 v16bf;
typedef __attribute__((ext_vector_type(8)))  __bf16 v8bf;
typedef __attribute__((ext_vector_type(8)))  float  v8f;
typedef __attribute__((ext_vector_type(4)))  float  v4f;

#define RNN_B 256
#define RNN_T 512
#define RNN_I 128
#define RNN_H 1024
#define RNN_O 128

#define KCAT   (RNN_I + RNN_H)     // 1152
#define KT_CAT (KCAT / 32)         // 36 k-steps in the fused scan GEMM

// LDS row strides (padded: row stride in dwords = 4 mod 64, so the 16
// row-parallel A-fragment gathers spread across all 64 LDS banks)
constexpr int CS = KCAT + 8;       // bf16 elements, scan activation buffer
constexpr int HS = RNN_H + 8;      // bf16 elements, out-kernel tile
constexpr int LS = RNN_O + 4;      // fp32 elements, logits

// ---------------------------------------------------------------------------
// Pack a row-major fp32 matrix src[N,K] into per-lane WMMA B-fragments (bf16)
// so that B[k][n] = src[n][k]  (the GEMM computes  A @ src^T).
// Fragment layout (wave32, 16x16x32 bf16 B operand):
//   lane<16 : n = nt*16+lane,    values k = {kt*32+0..7,  kt*32+16..23}
//   lane>=16: n = nt*16+lane-16, values k = {kt*32+8..15, kt*32+24..31}
// Tiles stored nt-major, kt-minor with a caller-chosen total KT and kt
// offset, so [W_in | Wrec] pack into one contiguous stream per N-tile.
// ---------------------------------------------------------------------------
__global__ void __launch_bounds__(32)
pack_b_kernel(const float* __restrict__ src, __bf16* __restrict__ dst,
              int K, int KTtot, int ktOff)
{
    const int kt   = blockIdx.x;       // 0 .. K/32-1
    const int nt   = blockIdx.y;
    const int lane = threadIdx.x;
    const int n    = nt * 16 + (lane & 15);
    const int hi   = lane >> 4;

    const float* srow = src + (size_t)n * K + kt * 32 + hi * 8;
    union { v16bf v; __bf16 e[16]; } u;
#pragma unroll
    for (int j = 0; j < 8; ++j) {
        u.e[j]     = (__bf16)srow[j];        // k = kt*32 + hi*8 + j
        u.e[j + 8] = (__bf16)srow[16 + j];   // k = kt*32 + 16 + hi*8 + j
    }
    ((v16bf*)dst)[((size_t)nt * KTtot + ktOff + kt) * 32 + lane] = u.v;
}

// ---------------------------------------------------------------------------
// Sequential scan: 16 WGs (one per 16-row batch tile), 16 waves of 32.
// Wave w owns H-columns [w*64, w*64+64) = 4 WMMA N-tiles; full K=1152
// reduction per step with ping-pong double-buffered B-fragment stream.
// ---------------------------------------------------------------------------
__global__ void __launch_bounds__(512)
rnn_scan_kernel(const float* __restrict__ x,
                const float* __restrict__ b_in,
                const __bf16* __restrict__ WcatP,
                float* __restrict__ hiddens)
{
    __shared__ __bf16 cat_s[16 * CS];   // [16 rows][x_t (128) | h (1024)]

    const int tid  = threadIdx.x;
    const int wave = tid >> 5;        // 0..15
    const int lane = tid & 31;
    const int row  = lane & 15;       // A-frag row / C-frag N index
    const int hi   = lane >> 4;
    const int b0   = blockIdx.x * 16;

    // h_0 = 0 (also clears x region; overwritten below)
    for (int i = tid; i < 16 * CS; i += 512) cat_s[i] = (__bf16)0.0f;

    float bias_in[4];
#pragma unroll
    for (int q = 0; q < 4; ++q) bias_in[q] = b_in[(wave * 4 + q) * 16 + row];

    // unified [W_in | Wrec] fragment stream base for this wave's 4 N-tiles
    const v16bf* Wc = (const v16bf*)WcatP +
                      (size_t)(wave * 4) * KT_CAT * 32 + lane;

    // each wave stages one x row (wave == batch row within tile)
    const float* xrow = x + (size_t)(b0 + wave) * RNN_T * RNN_I + lane * 4;

    // commit x(t=0) into the x region of cat_s
    {
        const v4f x0 = __builtin_nontemporal_load((const v4f*)(xrow));
        __bf16* d = cat_s + wave * CS + lane * 4;
        d[0] = (__bf16)x0.x; d[1] = (__bf16)x0.y;
        d[2] = (__bf16)x0.z; d[3] = (__bf16)x0.w;
    }
    __syncthreads();

    v4f xreg = {};   // holds x(t+1) during step t

    for (int t = 0; t < RNN_T; ++t) {
        // issue x(t+1) load now; it lands during the GEMM below
        if (t + 1 < RNN_T)
            xreg = __builtin_nontemporal_load(
                (const v4f*)(xrow + (size_t)(t + 1) * RNN_I));

        v8f acc[4];
#pragma unroll
        for (int q = 0; q < 4; ++q) {
#pragma unroll
            for (int r = 0; r < 8; ++r) acc[q][r] = bias_in[q];
        }

        // ---- fused GEMM: [16 x 1152] @ [W_in | Wrec]^T slice --------------
        {
            v16bf bp[4], bq[4];
#pragma unroll
            for (int q = 0; q < 4; ++q) bp[q] = Wc[(q * KT_CAT) * 32];

            for (int ks = 0; ks < KT_CAT; ks += 2) {
                // prefetch k+1 fragments
#pragma unroll
                for (int q = 0; q < 4; ++q)
                    bq[q] = Wc[(q * KT_CAT + ks + 1) * 32];
                {
                    union { v16bf v; v8bf h[2]; } a;
                    const __bf16* p = cat_s + row * CS + ks * 32 + hi * 8;
                    a.h[0] = *(const v8bf*)(p);
                    a.h[1] = *(const v8bf*)(p + 16);
#pragma unroll
                    for (int q = 0; q < 4; ++q)
                        acc[q] = __builtin_amdgcn_wmma_f32_16x16x32_bf16(
                            false, a.v, false, bp[q], (short)0, acc[q],
                            false, false);
                }
                // prefetch k+2 fragments while k+1 WMMAs run
                if (ks + 2 < KT_CAT) {
#pragma unroll
                    for (int q = 0; q < 4; ++q)
                        bp[q] = Wc[(q * KT_CAT + ks + 2) * 32];
                }
                {
                    union { v16bf v; v8bf h[2]; } a;
                    const __bf16* p = cat_s + row * CS + (ks + 1) * 32 + hi * 8;
                    a.h[0] = *(const v8bf*)(p);
                    a.h[1] = *(const v8bf*)(p + 16);
#pragma unroll
                    for (int q = 0; q < 4; ++q)
                        acc[q] = __builtin_amdgcn_wmma_f32_16x16x32_bf16(
                            false, a.v, false, bq[q], (short)0, acc[q],
                            false, false);
                }
            }
        }

        __syncthreads();   // all waves finished reading cat_s (x_t and h_{t-1})

        // ---- ReLU, write hiddens (fp32, streaming), refresh h region ------
#pragma unroll
        for (int q = 0; q < 4; ++q) {
            const int n = (wave * 4 + q) * 16 + row;
#pragma unroll
            for (int r = 0; r < 8; ++r) {
                float v = acc[q][r];
                v = v > 0.0f ? v : 0.0f;
                const int m = r + hi * 8;   // C layout: lane<16 -> M=r, else M=8+r
                __builtin_nontemporal_store(
                    v, &hiddens[((size_t)(b0 + m) * RNN_T + t) * RNN_H + n]);
                cat_s[m * CS + RNN_I + n] = (__bf16)v;
            }
        }
        // commit x(t+1) into the x region (junk on last iter; never read)
        {
            __bf16* d = cat_s + wave * CS + lane * 4;
            d[0] = (__bf16)xreg.x; d[1] = (__bf16)xreg.y;
            d[2] = (__bf16)xreg.z; d[3] = (__bf16)xreg.w;
        }
        __syncthreads();   // cat_s now holds [x_{t+1} | h_t]
    }
}

// ---------------------------------------------------------------------------
// Output projection + softmax, fully parallel over B*T rows.
// One WG per 16 rows of hiddens viewed as [B*T, H]; 8 waves, wave w owns
// output columns [w*16, w*16+16).
// ---------------------------------------------------------------------------
__global__ void __launch_bounds__(256)
rnn_out_kernel(const float* __restrict__ hiddens,
               const float* __restrict__ b_out,
               const __bf16* __restrict__ WoutP,
               float* __restrict__ outs)
{
    __shared__ __bf16 h_s[16 * HS];
    __shared__ float  l_s[16 * LS];
    __shared__ float  r_max[16];
    __shared__ float  r_inv[16];

    const int tid  = threadIdx.x;
    const int wave = tid >> 5;        // 0..7
    const int lane = tid & 31;
    const int row  = lane & 15;
    const int hi   = lane >> 4;
    const size_t R0 = (size_t)blockIdx.x * 16;   // flat row = b*T + t

    // ---- load 16x1024 fp32 tile (streaming), convert to bf16 in LDS -------
    {
        const int m  = tid >> 4;          // 0..15
        const int c0 = (tid & 15) * 4;    // 16 lanes cover 256B contiguously
        const float* src = hiddens + (R0 + m) * RNN_H + c0;
        __bf16* dst = h_s + m * HS + c0;
#pragma unroll
        for (int c = 0; c < 16; ++c) {
            const v4f v = __builtin_nontemporal_load((const v4f*)(src + c * 64));
            dst[c * 64 + 0] = (__bf16)v.x; dst[c * 64 + 1] = (__bf16)v.y;
            dst[c * 64 + 2] = (__bf16)v.z; dst[c * 64 + 3] = (__bf16)v.w;
        }
    }
    __syncthreads();

    // ---- logits: [16 x 1024] @ W_out^T (16-col slice per wave) ------------
    {
        const v16bf* Wo = (const v16bf*)WoutP +
                          (size_t)wave * (RNN_H / 32) * 32 + lane;
        v8f oacc;
        const float bo = b_out[wave * 16 + row];
#pragma unroll
        for (int r = 0; r < 8; ++r) oacc[r] = bo;

        v16bf bp = Wo[0], bq;
        for (int ks = 0; ks < RNN_H / 32; ks += 2) {
            bq = Wo[(ks + 1) * 32];
            {
                union { v16bf v; v8bf h[2]; } a;
                const __bf16* p = h_s + row * HS + ks * 32 + hi * 8;
                a.h[0] = *(const v8bf*)(p);
                a.h[1] = *(const v8bf*)(p + 16);
                oacc = __builtin_amdgcn_wmma_f32_16x16x32_bf16(
                    false, a.v, false, bp, (short)0, oacc, false, false);
            }
            if (ks + 2 < RNN_H / 32) bp = Wo[(ks + 2) * 32];
            {
                union { v16bf v; v8bf h[2]; } a;
                const __bf16* p = h_s + row * HS + (ks + 1) * 32 + hi * 8;
                a.h[0] = *(const v8bf*)(p);
                a.h[1] = *(const v8bf*)(p + 16);
                oacc = __builtin_amdgcn_wmma_f32_16x16x32_bf16(
                    false, a.v, false, bq, (short)0, oacc, false, false);
            }
        }
        const int n = wave * 16 + row;
#pragma unroll
        for (int r = 0; r < 8; ++r) l_s[(r + hi * 8) * LS + n] = oacc[r];
    }
    __syncthreads();

    // ---- row-wise softmax stats -------------------------------------------
    if (tid < 16) {
        float mx = -3.0e38f;
        for (int j = 0; j < RNN_O; ++j) mx = fmaxf(mx, l_s[tid * LS + j]);
        float s = 0.0f;
        for (int j = 0; j < RNN_O; ++j) s += __expf(l_s[tid * LS + j] - mx);
        r_max[tid] = mx;
        r_inv[tid] = 1.0f / s;
    }
    __syncthreads();

    // ---- write outs: 16x128 fp32 (streaming), 8 values per thread ---------
    {
        const int m  = tid >> 4;
        const int j0 = (tid & 15) * 8;
        const float mx = r_max[m], is = r_inv[m];
        float* dst = outs + (R0 + m) * RNN_O + j0;
        v4f o0, o1;
        o0.x = __expf(l_s[m * LS + j0 + 0] - mx) * is;
        o0.y = __expf(l_s[m * LS + j0 + 1] - mx) * is;
        o0.z = __expf(l_s[m * LS + j0 + 2] - mx) * is;
        o0.w = __expf(l_s[m * LS + j0 + 3] - mx) * is;
        o1.x = __expf(l_s[m * LS + j0 + 4] - mx) * is;
        o1.y = __expf(l_s[m * LS + j0 + 5] - mx) * is;
        o1.z = __expf(l_s[m * LS + j0 + 6] - mx) * is;
        o1.w = __expf(l_s[m * LS + j0 + 7] - mx) * is;
        __builtin_nontemporal_store(o0, (v4f*)(dst));
        __builtin_nontemporal_store(o1, (v4f*)(dst + 4));
    }
}

// ---------------------------------------------------------------------------
extern "C" void kernel_launch(void* const* d_in, const int* in_sizes, int n_in,
                              void* d_out, int out_size, void* d_ws, size_t ws_size,
                              hipStream_t stream)
{
    (void)in_sizes; (void)n_in; (void)out_size; (void)ws_size;

    const float* x     = (const float*)d_in[0];
    const float* W_in  = (const float*)d_in[1];
    const float* b_in  = (const float*)d_in[2];
    const float* Wrec  = (const float*)d_in[3];
    const float* W_out = (const float*)d_in[4];
    const float* b_out = (const float*)d_in[5];

    // workspace: packed bf16 fragment streams (2.5 MB total, L2-resident)
    __bf16* WcatP = (__bf16*)d_ws;                                  // 2.25 MB
    __bf16* WoutP = WcatP + (size_t)(RNN_H / 16) * KT_CAT * 32 * 16;

    // [W_in | Wrec] -> one unified stream, KT = 36 per N-tile
    pack_b_kernel<<<dim3(RNN_I / 32, RNN_H / 16), 32, 0, stream>>>(
        W_in, WcatP, RNN_I, KT_CAT, 0);
    pack_b_kernel<<<dim3(RNN_H / 32, RNN_H / 16), 32, 0, stream>>>(
        Wrec, WcatP, RNN_H, KT_CAT, RNN_I / 32);
    // W_out -> its own stream, KT = 32 per N-tile
    pack_b_kernel<<<dim3(RNN_H / 32, RNN_O / 16), 32, 0, stream>>>(
        W_out, WoutP, RNN_H, RNN_H / 32, 0);

    float* hiddens = (float*)d_out;
    float* outs    = hiddens + (size_t)RNN_B * RNN_T * RNN_H;

    // sequential scan: 16 batch tiles x 512 threads (16 waves)
    rnn_scan_kernel<<<dim3(RNN_B / 16), 512, 0, stream>>>(
        x, b_in, WcatP, hiddens);

    // deferred output projection + softmax: full-chip parallel over B*T rows
    rnn_out_kernel<<<dim3((RNN_B * RNN_T) / 16), 256, 0, stream>>>(
        hiddens, b_out, WoutP, outs);
}
